// ComplexMoELayer_48000554500926
// MI455X (gfx1250) — matrix operand order repaired and checked by reading, compiled, not run on hardware
//
#include <hip/hip_runtime.h>
#include <math.h>

#define NTOK 2048   // B*S
#define E_ 8
#define D_ 512
#define H_ 2048

typedef __attribute__((ext_vector_type(16))) __bf16 v16bf;
typedef __attribute__((ext_vector_type(2)))  __bf16 bf16x2;
typedef __attribute__((ext_vector_type(8)))  float  v8f;

union FragBF { v16bf v; unsigned int u[8]; };
union FragF  { v8f v;   float f[8]; };

// packed fp32 -> bf16x2 (RNE) -- lowers to the packed cvt path
__device__ __forceinline__ unsigned int pack2bf(float a, float b) {
  union { bf16x2 v; unsigned int u; } c;
  c.v.x = (__bf16)a;
  c.v.y = (__bf16)b;
  return c.u;
}

// ---------------- gating: amp/phase -> scores -> softmax -> top-1 ----------------
__global__ void gate_kernel(const float* __restrict__ xr, const float* __restrict__ xi,
                            const float* __restrict__ gW, const float* __restrict__ gb,
                            int* __restrict__ topk_idx, float* __restrict__ topk_w,
                            int* __restrict__ counts, int* __restrict__ elist) {
  __shared__ float sred[256 * E_];
  const int t = blockIdx.x;
  const int tid = threadIdx.x;
  float part[E_];
  for (int e = 0; e < E_; ++e) part[e] = 0.f;
  for (int d = tid; d < D_; d += 256) {
    float r = xr[(size_t)t * D_ + d];
    float i = xi[(size_t)t * D_ + d];
    float amp = sqrtf(r * r + i * i);
    float ph  = atan2f(i, r);
    const float* wA = gW + (size_t)d * E_;
    const float* wP = gW + (size_t)(d + D_) * E_;
    for (int e = 0; e < E_; ++e) part[e] += amp * wA[e] + ph * wP[e];
  }
  for (int e = 0; e < E_; ++e) sred[tid * E_ + e] = part[e];
  __syncthreads();
  for (int stride = 128; stride > 0; stride >>= 1) {
    if (tid < stride)
      for (int e = 0; e < E_; ++e)
        sred[tid * E_ + e] += sred[(tid + stride) * E_ + e];
    __syncthreads();
  }
  if (tid == 0) {
    float s[E_]; float mx = -1e30f; int arg = 0;
    for (int e = 0; e < E_; ++e) {
      s[e] = sred[e] + gb[e];
      if (s[e] > mx) { mx = s[e]; arg = e; }   // first-max, matches jnp.argmax
    }
    float sum = 0.f;
    for (int e = 0; e < E_; ++e) { s[e] = expf(s[e] - mx); sum += s[e]; }
    topk_idx[t] = arg;
    topk_w[t]   = s[arg] / sum;
    int slot = atomicAdd(&counts[arg], 1);
    elist[arg * NTOK + slot] = t;
  }
}

// ---------------- complex FFN layer via bf16 WMMA ----------------
// Tile: 64 tokens x 128 out-cols per block (8 waves, each wave = one 16-col strip,
// 4 row-subtiles). Per K=32 chunk: all A frags loaded first, then a 16-WMMA burst
// with no VALU interleaved and alternating accumulators (no hazard NOPs needed).
template<int K_DIM, int N_DIM, bool LAYER2>
__global__ void __launch_bounds__(256)
ffn_kernel(const float* __restrict__ Ar_src, const float* __restrict__ Ai_src,
           const float* __restrict__ Wr, const float* __restrict__ Wi,
           const float* __restrict__ br, const float* __restrict__ bi,
           const float* __restrict__ modb, const float* __restrict__ tw,
           const int* __restrict__ counts, const int* __restrict__ elist,
           float* __restrict__ Or, float* __restrict__ Oi) {
  constexpr int MT = 64;    // tokens per block
  constexpr int NT = 128;   // output cols per block
  constexpr int KT = 32;    // K per WMMA chunk
  constexpr int RT = 4;     // 16-row subtiles per wave
  constexpr int LP = 34;    // LDS row pitch (ushort): stride 68B -> bank delta 17, conflict-free

  __shared__ unsigned short sAr[MT * LP];
  __shared__ unsigned short sAi[MT * LP];
  __shared__ unsigned short sBr[NT * LP];   // stored transposed: [col][k]
  __shared__ unsigned short sBi[NT * LP];
  __shared__ int sTok[MT];

  const int e  = blockIdx.x;
  const int mt = blockIdx.y;
  const int nb = blockIdx.z * NT;
  const int cnt = counts[e];
  if (mt * MT >= cnt) return;   // idle tile for this expert: cheap early exit

  const int tid   = threadIdx.x;
  const int wave  = tid >> 5;
  const int lane  = tid & 31;
  const int lhalf = lane & 15;
  const bool hi   = lane >= 16;

  if (tid < MT) {
    int slot = mt * MT + tid;
    if (slot >= cnt) slot = cnt - 1;       // clamp; padded rows masked at store
    sTok[tid] = elist[e * NTOK + slot];
  }
  __syncthreads();

  const float* We_r = Wr + (size_t)e * K_DIM * N_DIM;
  const float* We_i = Wi + (size_t)e * K_DIM * N_DIM;

  v8f accR[RT], accI[RT];
  const v8f vzero = {0.f, 0.f, 0.f, 0.f, 0.f, 0.f, 0.f, 0.f};
  for (int r = 0; r < RT; ++r) { accR[r] = vzero; accI[r] = vzero; }

  for (int kb = 0; kb < K_DIM / KT; ++kb) {
    __syncthreads();  // protect LDS from previous iteration's fragment reads

    // ---- stage A (gathered token rows), fp32 -> packed bf16x2 ----
    #pragma unroll
    for (int i = 0; i < (MT * KT / 2) / 256; ++i) {   // 4 iters
      int pidx = tid + i * 256;
      int m = pidx >> 4, k = (pidx & 15) * 2;
      size_t g = (size_t)sTok[m] * K_DIM + (size_t)kb * KT + k;
      float2 ar = *(const float2*)(Ar_src + g);
      float2 ai = *(const float2*)(Ai_src + g);
      *(unsigned int*)&sAr[m * LP + k] = pack2bf(ar.x, ar.y);
      *(unsigned int*)&sAi[m * LP + k] = pack2bf(ai.x, ai.y);
    }
    // ---- stage B (weights), fp32 -> bf16, transposed [col][k] ----
    if (kb + 1 < K_DIM / KT) {   // software prefetch of next weight chunk
      size_t pg = (size_t)(kb + 1) * KT * N_DIM + nb + (tid & 127);
      __builtin_prefetch(&We_r[pg], 0, 1);
      __builtin_prefetch(&We_i[pg], 0, 1);
    }
    #pragma unroll
    for (int i = 0; i < (KT / 2 * NT) / 256; ++i) {   // 8 iters
      int pidx = tid + i * 256;
      int c = pidx & 127, k = (pidx >> 7) * 2;
      size_t g0 = (size_t)(kb * KT + k) * N_DIM + nb + c;
      *(unsigned int*)&sBr[c * LP + k] = pack2bf(We_r[g0], We_r[g0 + N_DIM]);
      *(unsigned int*)&sBi[c * LP + k] = pack2bf(We_i[g0], We_i[g0 + N_DIM]);
    }
    __syncthreads();

    // ---- B fragments (32x16 KxN bf16): lane n=L%16; K=j (lanes 0-15), K=16+j (16-31)
    FragBF fBr, fBi;
    {
      const unsigned short* pr = &sBr[(wave * 16 + lhalf) * LP];
      const unsigned short* pi = &sBi[(wave * 16 + lhalf) * LP];
      const int kb0 = hi ? 16 : 0;
      #pragma unroll
      for (int t8 = 0; t8 < 8; ++t8) {
        int k = kb0 + 2 * t8;
        fBr.u[t8] = *(const unsigned int*)(pr + k);
        fBi.u[t8] = *(const unsigned int*)(pi + k);
      }
    }

    // ---- all A fragments up front (16x32 MxK bf16, K=j (lanes 0-15), K=16+j (16-31), same order as B) ----
    FragBF fAr[RT], fAi[RT], fAn[RT];
    #pragma unroll
    for (int rt = 0; rt < RT; ++rt) {
      const unsigned short* pr = &sAr[(rt * 16 + lhalf) * LP];
      const unsigned short* pi = &sAi[(rt * 16 + lhalf) * LP];
      #pragma unroll
      for (int t8 = 0; t8 < 8; ++t8) {
        int j = 2 * t8;
        int k = (hi ? 16 : 0) + j;
        fAr[rt].u[t8] = *(const unsigned int*)(pr + k);
        unsigned int ui = *(const unsigned int*)(pi + k);
        fAi[rt].u[t8] = ui;
        fAn[rt].u[t8] = ui ^ 0x80008000u;   // -Ai (NEG bits are C-only for bf16 WMMA)
      }
    }

    // ---- 16-WMMA burst: consecutive ops hit different accumulators; sources
    //      are read-only for the whole burst -> no D->A/B RAW, no VALU WAR ----
    #pragma unroll
    for (int rt = 0; rt < RT; ++rt)   // acc_r += Ar*Br
      accR[rt] = __builtin_amdgcn_wmma_f32_16x16x32_bf16(false, fAr[rt].v, false, fBr.v, (short)0, accR[rt], false, false);
    #pragma unroll
    for (int rt = 0; rt < RT; ++rt)   // acc_i += Ar*Bi
      accI[rt] = __builtin_amdgcn_wmma_f32_16x16x32_bf16(false, fAr[rt].v, false, fBi.v, (short)0, accI[rt], false, false);
    #pragma unroll
    for (int rt = 0; rt < RT; ++rt)   // acc_r += (-Ai)*Bi
      accR[rt] = __builtin_amdgcn_wmma_f32_16x16x32_bf16(false, fAn[rt].v, false, fBi.v, (short)0, accR[rt], false, false);
    #pragma unroll
    for (int rt = 0; rt < RT; ++rt)   // acc_i += Ai*Br
      accI[rt] = __builtin_amdgcn_wmma_f32_16x16x32_bf16(false, fAi[rt].v, false, fBr.v, (short)0, accI[rt], false, false);
  }

  // ---- fused epilogue on C fragments (lane n=L%16; VGPR v -> row v + 8*(L>=16)) ----
  const int colg = nb + wave * 16 + lhalf;
  const float biasr = br[(size_t)e * N_DIM + colg];
  const float biasi = bi[(size_t)e * N_DIM + colg];
  #pragma unroll
  for (int rt = 0; rt < RT; ++rt) {
    FragF fr, fi;
    fr.v = accR[rt]; fi.v = accI[rt];
    #pragma unroll
    for (int v = 0; v < 8; ++v) {
      int m = rt * 16 + v + (hi ? 8 : 0);
      int slot = mt * MT + m;
      if (slot >= cnt) continue;     // padded rows
      int tok = sTok[m];
      float hr = fr.f[v] + biasr;
      float hx = fi.f[v] + biasi;
      if constexpr (!LAYER2) {
        // ComplexModReLU: relu(|z|+b)*z/(|z|+eps), |z|=sqrt(r^2+i^2+eps)
        float a = sqrtf(hr * hr + hx * hx + 1e-10f);
        float s = fmaxf(a + modb[(size_t)e * N_DIM + colg], 0.f) / (a + 1e-10f);
        Or[(size_t)tok * N_DIM + colg] = hr * s;
        Oi[(size_t)tok * N_DIM + colg] = hx * s;
      } else {
        float w = tw[tok];            // top-1 softmax prob
        Or[(size_t)tok * N_DIM + colg] = hr * w;
        Oi[(size_t)tok * N_DIM + colg] = hx * w;
      }
    }
  }
}

extern "C" void kernel_launch(void* const* d_in, const int* in_sizes, int n_in,
                              void* d_out, int out_size, void* d_ws, size_t ws_size,
                              hipStream_t stream) {
  const float* xr   = (const float*)d_in[0];
  const float* xi   = (const float*)d_in[1];
  const float* gW   = (const float*)d_in[2];
  const float* gb   = (const float*)d_in[3];
  const float* W1r  = (const float*)d_in[4];
  const float* W1i  = (const float*)d_in[5];
  const float* b1r  = (const float*)d_in[6];
  const float* b1i  = (const float*)d_in[7];
  const float* modb = (const float*)d_in[8];
  const float* W2r  = (const float*)d_in[9];
  const float* W2i  = (const float*)d_in[10];
  const float* b2r  = (const float*)d_in[11];
  const float* b2i  = (const float*)d_in[12];
  float* out = (float*)d_out;

  // workspace carve-up
  char* ws = (char*)d_ws;
  int*   counts = (int*)ws;                  ws += 256;
  int*   tidx   = (int*)ws;                  ws += NTOK * sizeof(int);
  float* tw     = (float*)ws;                ws += NTOK * sizeof(float);
  int*   elist  = (int*)ws;                  ws += (size_t)E_ * NTOK * sizeof(int);
  float* hr     = (float*)ws;                ws += (size_t)NTOK * H_ * sizeof(float);
  float* hi     = (float*)ws;

  hipMemsetAsync(counts, 0, E_ * sizeof(int), stream);  // graph-capturable

  gate_kernel<<<NTOK, 256, 0, stream>>>(xr, xi, gW, gb, tidx, tw, counts, elist);

  // layer 1: [tok, D] x [D, H] complex, + ModReLU -> h in workspace
  ffn_kernel<D_, H_, false><<<dim3(E_, NTOK / 64, H_ / 128), 256, 0, stream>>>(
      xr, xi, W1r, W1i, b1r, b1i, modb, nullptr, counts, elist, hr, hi);

  // layer 2: [tok, H] x [H, D] complex, + bias + prob scale -> scatter to out
  ffn_kernel<H_, D_, true><<<dim3(E_, NTOK / 64, D_ / 128), 256, 0, stream>>>(
      hr, hi, W2r, W2i, b2r, b2i, nullptr, tw, counts, elist,
      out, out + (size_t)NTOK * D_);
}
